// AdditiveAttention_69595650064832
// MI455X (gfx1250) — matrix operand order repaired and checked by reading, not compiled, not run on hardware
//
#include <hip/hip_runtime.h>
#include <hip/hip_bf16.h>

namespace {
constexpr int B_  = 8;
constexpr int LQ_ = 256;
constexpr int LK_ = 256;
constexpr int DQ_ = 512;
constexpr int DV_ = 512;
constexpr int H_  = 256;
constexpr int PADH = 260;   // LDS row stride over h (multiple of 4 floats, %64 == 4 -> conflict-free)
constexpr int PADK = 260;   // LDS row stride over lk
}

typedef __attribute__((ext_vector_type(16))) __bf16 v16bf;
typedef __attribute__((ext_vector_type(8)))  float  v8f;

#if __has_builtin(__builtin_amdgcn_tanhf)
__device__ __forceinline__ float fast_tanh(float x) { return __builtin_amdgcn_tanhf(x); }
#else
__device__ __forceinline__ float fast_tanh(float x) { return tanhf(x); }
#endif

// -------------------------------------------------------------------------
// Projection GEMM: out[row, h] = sum_d X[row, d] * W[h, d] + bias[h] (+bias2[h])
// One wave per 16x16 output tile; bf16 WMMA 16x16x32, f32 accumulate.
// A-fragment (16x32, 16-bit): lane m=lane&15, half=lane>>4,
//   vgpr j holds K = half*16 + 2j  (and +1)
// B-fragment (32x16, 16-bit): lane n=lane&15, halves hold K 0-15 / 16-31,
//   vgpr j holds K = half*16 + 2j (and +1); B[k][n] = W[h0+n][k]
// C/D (16x16 f32): vgpr r -> M = r + 8*half, N = lane&15
// -------------------------------------------------------------------------
__global__ __launch_bounds__(128)
void proj_bf16_wmma(const float* __restrict__ X, const float* __restrict__ W,
                    const float* __restrict__ bias, const float* __restrict__ bias2,
                    float* __restrict__ out, int L, int D, int addBias2)
{
    const int lane = threadIdx.x & 31;
    const int warp = threadIdx.x >> 5;
    const int tilesH = H_ / 16;
    const int tiles  = (B_ * L / 16) * tilesH;
    const int tile   = blockIdx.x * 4 + warp;
    if (tile >= tiles) return;

    const int ht   = tile % tilesH;
    const int rt   = tile / tilesH;
    const int row0 = rt * 16;
    const int h0   = ht * 16;
    const int m16  = lane & 15;
    const int half = lane >> 4;

    const float* Arow = X + (size_t)(row0 + m16) * D;  // A row m16 (two lanes per row, K-split)
    const float* Brow = W + (size_t)(h0  + m16) * D;   // B column n = m16 is W row h0+n

    v8f acc = {0.f, 0.f, 0.f, 0.f, 0.f, 0.f, 0.f, 0.f};
#pragma unroll 2
    for (int k0 = 0; k0 < D; k0 += 32) {
        v16bf a, b;
        const int koff = half * 16;
#pragma unroll
        for (int j = 0; j < 8; ++j) {
            const int kb = half * 16 + 2 * j;
            a[2*j]   = (__bf16)Arow[k0 + kb];
            a[2*j+1] = (__bf16)Arow[k0 + kb + 1];
            b[2*j]   = (__bf16)Brow[k0 + koff + 2*j];
            b[2*j+1] = (__bf16)Brow[k0 + koff + 2*j + 1];
        }
        acc = __builtin_amdgcn_wmma_f32_16x16x32_bf16(false, a, false, b,
                                                      (short)0, acc, false, false);
    }

    const float bsum = bias[h0 + m16] + (addBias2 ? bias2[h0 + m16] : 0.0f);
#pragma unroll
    for (int r = 0; r < 8; ++r) {
        const int m = r + 8 * half;
        out[(size_t)(row0 + m) * H_ + h0 + m16] = acc[r] + bsum;
    }
}

// -------------------------------------------------------------------------
// Energies + masked softmax + context GEMM.
// One workgroup (256 threads, 8 waves) per (b, 16-row q tile).
// -------------------------------------------------------------------------
__global__ __launch_bounds__(256)
void attn_energy_softmax_ctx(const float* __restrict__ qp,   // [B,LQ,H] (has bq + b_param)
                             const float* __restrict__ kp,   // [B,LK,H] (has bk)
                             const float* __restrict__ Vp,   // [B,LK,DV]
                             const int*   __restrict__ mask, // [B,LK]
                             const float* __restrict__ vpar, // [H]
                             float* __restrict__ ctx,        // [B,LQ,DV]
                             float* __restrict__ attn_out)   // [B,LQ,LK]
{
    __shared__ __align__(16) float s_q[16 * PADH];
    __shared__ __align__(16) float s_k[16 * PADH];
    __shared__ __align__(16) float s_e[16 * PADK];
    __shared__ __align__(16) float s_v[H_];
    __shared__ float s_red[16 * 16];
    __shared__ int   s_mask[LK_];

    const int tid = threadIdx.x;
    const int b   = blockIdx.x / (LQ_ / 16);
    const int lq0 = (blockIdx.x % (LQ_ / 16)) * 16;

    // stage q tile, v_param, mask
    for (int i = tid; i < 16 * H_; i += 256) {
        const int r = i >> 8, h = i & 255;
        s_q[r * PADH + h] = qp[((size_t)(b * LQ_ + lq0 + r)) * H_ + h];
    }
    for (int i = tid; i < H_; i += 256)  s_v[i] = vpar[i];
    for (int i = tid; i < LK_; i += 256) s_mask[i] = mask[b * LK_ + i];

    const int elq = tid >> 4;   // 0..15 : q row within tile
    const int elk = tid & 15;   // 0..15 : k row within chunk

    // ---- energies: e = sum_h v_h * tanh(q_h + k_h) ----
    const float* kpb = kp + (size_t)b * LK_ * H_;
    for (int kc = 0; kc < LK_ / 16; ++kc) {
        __syncthreads();
        for (int i = tid; i < 16 * H_; i += 256) {
            const int r = i >> 8, h = i & 255;
            s_k[r * PADH + h] = kpb[(size_t)(kc * 16 + r) * H_ + h];
        }
        // prefetch next chunk (16 rows x 256 f32 = 16 KB) into cache while
        // the tanh loop below runs on the TRANS pipe (global_prefetch_b8)
        if (kc + 1 < LK_ / 16 && tid < 128)
            __builtin_prefetch(kpb + (size_t)(kc + 1) * 16 * H_ + tid * 32, 0, 3);
        __syncthreads();

        const float* qrow = &s_q[elq * PADH];
        const float* krow = &s_k[elk * PADH];
        float e = 0.0f;
#pragma unroll 4
        for (int h = 0; h < H_; h += 4) {
            const float4 qv = *(const float4*)&qrow[h];
            const float4 kv = *(const float4*)&krow[h];
            const float4 vv = *(const float4*)&s_v[h];
            e += vv.x * fast_tanh(qv.x + kv.x);
            e += vv.y * fast_tanh(qv.y + kv.y);
            e += vv.z * fast_tanh(qv.z + kv.z);
            e += vv.w * fast_tanh(qv.w + kv.w);
        }
        const int lk = kc * 16 + elk;
        if (s_mask[lk] == 0) e = -__builtin_inff();
        s_e[elq * PADK + lk] = e;
    }
    __syncthreads();

    // ---- softmax over lk (two-stage reductions in LDS) ----
    float pmax = -__builtin_inff();
    for (int k = elk; k < LK_; k += 16) pmax = fmaxf(pmax, s_e[elq * PADK + k]);
    s_red[elq * 16 + elk] = pmax;
    __syncthreads();
    if (elk == 0) {
        float mm = -__builtin_inff();
        for (int i = 0; i < 16; ++i) mm = fmaxf(mm, s_red[elq * 16 + i]);
        s_red[elq * 16] = mm;
    }
    __syncthreads();
    const float rowmax = s_red[elq * 16];
    __syncthreads();

    float psum = 0.0f;
    for (int k = elk; k < LK_; k += 16) {
        const float ex = __expf(s_e[elq * PADK + k] - rowmax);
        s_e[elq * PADK + k] = ex;
        psum += ex;
    }
    s_red[elq * 16 + elk] = psum;
    __syncthreads();
    if (elk == 0) {
        float ss = 0.0f;
        for (int i = 0; i < 16; ++i) ss += s_red[elq * 16 + i];
        s_red[elq * 16] = ss;
    }
    __syncthreads();
    const float inv = 1.0f / s_red[elq * 16];
    for (int k = elk; k < LK_; k += 16) {
        const float a = s_e[elq * PADK + k] * inv;
        s_e[elq * PADK + k] = a;
        attn_out[((size_t)(b * LQ_ + lq0 + elq)) * LK_ + k] = a;
    }
    __syncthreads();

    // ---- context GEMM: ctx[16 x 512] = attn[16 x 256] * V[256 x 512] ----
    const int lane = tid & 31;
    const int warp = tid >> 5;      // 8 waves x 4 n-tiles = 32 tiles of 16 cols
    const int m16  = lane & 15;
    const int half = lane >> 4;
    const float* Vb = Vp + (size_t)b * LK_ * DV_;

    v8f acc[4];
#pragma unroll
    for (int t = 0; t < 4; ++t) acc[t] = (v8f){0.f,0.f,0.f,0.f,0.f,0.f,0.f,0.f};

    for (int k0 = 0; k0 < LK_; k0 += 32) {
        const int koff = half * 16;
        // prefetch next K-block of this wave's V columns
        if (k0 + 32 < LK_)
            __builtin_prefetch(Vb + (size_t)(k0 + 32 + koff + (m16 & 7) * 2) * DV_
                                   + warp * 64, 0, 3);
        v16bf a;
#pragma unroll
        for (int j = 0; j < 8; ++j) {
            const int kb = half * 16 + 2 * j;
            a[2*j]   = (__bf16)s_e[m16 * PADK + k0 + kb];
            a[2*j+1] = (__bf16)s_e[m16 * PADK + k0 + kb + 1];
        }
#pragma unroll
        for (int t = 0; t < 4; ++t) {
            const int dv0 = (warp * 4 + t) * 16;
            v16bf bb;
#pragma unroll
            for (int j = 0; j < 8; ++j) {
                bb[2*j]   = (__bf16)Vb[(size_t)(k0 + koff + 2*j)     * DV_ + dv0 + m16];
                bb[2*j+1] = (__bf16)Vb[(size_t)(k0 + koff + 2*j + 1) * DV_ + dv0 + m16];
            }
            acc[t] = __builtin_amdgcn_wmma_f32_16x16x32_bf16(false, a, false, bb,
                                                             (short)0, acc[t], false, false);
        }
    }
#pragma unroll
    for (int t = 0; t < 4; ++t) {
        const int dv0 = (warp * 4 + t) * 16;
#pragma unroll
        for (int r = 0; r < 8; ++r) {
            const int m = r + 8 * half;
            ctx[((size_t)(b * LQ_ + lq0 + m)) * DV_ + dv0 + m16] = acc[t][r];
        }
    }
}

// -------------------------------------------------------------------------
extern "C" void kernel_launch(void* const* d_in, const int* in_sizes, int n_in,
                              void* d_out, int out_size, void* d_ws, size_t ws_size,
                              hipStream_t stream) {
    (void)in_sizes; (void)n_in; (void)out_size; (void)ws_size;

    const float* Q    = (const float*)d_in[0];
    const float* K    = (const float*)d_in[1];
    const float* V    = (const float*)d_in[2];
    const int*   mask = (const int*)  d_in[3];
    const float* Wq   = (const float*)d_in[4];
    const float* bq   = (const float*)d_in[5];
    const float* Wk   = (const float*)d_in[6];
    const float* bk   = (const float*)d_in[7];
    const float* vpar = (const float*)d_in[8];
    const float* bpar = (const float*)d_in[9];

    float* qp = (float*)d_ws;                       // [B,LQ,H]  2 MB
    float* kp = qp + (size_t)B_ * LQ_ * H_;         // [B,LK,H]  2 MB

    float* ctx  = (float*)d_out;                    // [B,LQ,DV]
    float* attn = ctx + (size_t)B_ * LQ_ * DV_;     // [B,LQ,LK]

    const int tiles  = (B_ * LQ_ / 16) * (H_ / 16);
    const int blocks = (tiles + 3) / 4;
    // q projection folds bq + b_param so the energy loop is tanh(q+k)*v only
    proj_bf16_wmma<<<blocks, 128, 0, stream>>>(Q, Wq, bq, bpar, qp, LQ_, DQ_, 1);
    proj_bf16_wmma<<<blocks, 128, 0, stream>>>(K, Wk, bk, bpar, kp, LK_, DQ_, 0);

    attn_energy_softmax_ctx<<<B_ * (LQ_ / 16), 256, 0, stream>>>(
        qp, kp, V, mask, vpar, ctx, attn);
}
